// TernaryDecoder_77824807404254
// MI455X (gfx1250) — matrix operand order repaired and checked by reading, not compiled, not run on hardware
//
#include <hip/hip_runtime.h>

// ---------- vector types for WMMA ----------
typedef __attribute__((ext_vector_type(16))) __bf16         v16bf;
typedef __attribute__((ext_vector_type(8)))  float          v8f;
typedef __attribute__((ext_vector_type(8)))  unsigned short v8u;
typedef __attribute__((ext_vector_type(16))) unsigned short v16u;
typedef __attribute__((ext_vector_type(4)))  int            v4i;

// Async global->LDS path (CDNA5). Gated so a missing builtin falls back to the
// synchronous global->VGPR->ds_store path and the TU still compiles.
#if defined(__HIP_DEVICE_COMPILE__) && defined(__has_builtin)
#if __has_builtin(__builtin_amdgcn_global_load_async_to_lds_b128)
#define HAVE_ASYNC_LDS 1
#endif
#endif
#ifndef HAVE_ASYNC_LDS
#define HAVE_ASYNC_LDS 0
#endif

#if HAVE_ASYNC_LDS
#if __has_builtin(__builtin_amdgcn_s_wait_asynccnt)
#define ASYNC_WAIT(n) __builtin_amdgcn_s_wait_asynccnt(n)
#else
#define ASYNC_WAIT(n) asm volatile("s_wait_asynccnt %0" ::"i"(n) : "memory")
#endif
// builtin signature (from clang diagnostic): param0 is int4 in AS1 (global).
typedef __attribute__((address_space(1))) void* gas1v_t;
typedef __attribute__((address_space(3))) void* las3v_t;
typedef __attribute__((address_space(1))) v4i*  gas1_t;
typedef __attribute__((address_space(3))) v4i*  las3_t;
#define GCAST(p) ((gas1_t)(gas1v_t)(void*)(p))
#define LCAST(p) ((las3_t)(las3v_t)(void*)(p))
#endif

// round-to-nearest-even f32 -> bf16 (raw u16). Ternary values are exact.
__device__ __forceinline__ unsigned short f32_to_bf16(float f) {
  unsigned int u = __builtin_bit_cast(unsigned int, f);
  u += 0x7FFFu + ((u >> 16) & 1u);
  return (unsigned short)(u >> 16);
}

// ---------- kernel 1: activations f32 -> bf16 ----------
__global__ __launch_bounds__(256)
void cvt_f32_bf16(const float* __restrict__ x, unsigned short* __restrict__ y,
                  size_t n) {
  size_t i = (size_t)blockIdx.x * blockDim.x + threadIdx.x;
  size_t stride = (size_t)gridDim.x * blockDim.x;
  for (; i < n; i += stride) y[i] = f32_to_bf16(x[i]);
}

// ---------- kernel 2: per-row ternary quantization -> bf16 ----------
__global__ __launch_bounds__(256)
void ternary_quant_rows(const float* __restrict__ W,
                        unsigned short* __restrict__ Wq, int K) {
  __shared__ float red[256];
  const int row = blockIdx.x;
  const int tid = threadIdx.x;
  const float* w = W + (size_t)row * K;
  float s = 0.f;
  for (int k = tid; k < K; k += 256) s += fabsf(w[k]);
  red[tid] = s;
  __syncthreads();
  for (int off = 128; off > 0; off >>= 1) {
    if (tid < off) red[tid] += red[tid + off];
    __syncthreads();
  }
  const float thr = 0.7f * red[0] / (float)K;
  unsigned short* q = Wq + (size_t)row * K;
  for (int k = tid; k < K; k += 256) {
    float v = w[k];
    float t = (v > thr) ? 1.f : ((v < -thr) ? -1.f : 0.f);
    q[k] = f32_to_bf16(t);
  }
}

// ---------- kernel 3: tiled bf16 WMMA GEMM ----------
// C[M,N] = A[M,K] * W[N,K]^T + bias, optional relu.
// Workgroup: 256 threads (8 waves). Tile: BM=256, BN=64, BK=32.
// Wave w computes rows [32w, 32w+32) x 64 cols: 2 A frags x 4 B frags = 8 WMMA.
// Double-buffered LDS fed by async global->LDS loads (ASYNCcnt), last K-step
// peeled so the steady-state loop has no conditional waits.
#define AS 40  // LDS row stride (elems): keeps 16B alignment, depads banks
#define BS 40

// per-thread issue of one 256x32 A tile + 64x32 B tile (5 x b128 each)
__device__ __forceinline__ void issue_tile(const unsigned short* __restrict__ A,
                                           const unsigned short* __restrict__ W,
                                           unsigned short* As,
                                           unsigned short* Bs, size_t mBase,
                                           size_t nBase, int K, int k0,
                                           int tid) {
#if HAVE_ASYNC_LDS
#pragma unroll
  for (int i = 0; i < 4; ++i) {
    const int vecIdx = tid + i * 256;  // 1024 x 16B = 256x32 A tile
    const int row = vecIdx >> 2, cv = vecIdx & 3;
    __builtin_amdgcn_global_load_async_to_lds_b128(
        GCAST(&A[(mBase + row) * K + k0 + cv * 8]),
        LCAST(&As[row * AS + cv * 8]), 0, 0);
  }
  {
    const int row = tid >> 2, cv = tid & 3;  // 256 x 16B = 64x32 B tile
    __builtin_amdgcn_global_load_async_to_lds_b128(
        GCAST(&W[(nBase + row) * K + k0 + cv * 8]),
        LCAST(&Bs[row * BS + cv * 8]), 0, 0);
  }
#else
#pragma unroll
  for (int i = 0; i < 4; ++i) {
    const int vecIdx = tid + i * 256;
    const int row = vecIdx >> 2, cv = vecIdx & 3;
    *(v8u*)&As[row * AS + cv * 8] =
        *(const v8u*)&A[(mBase + row) * K + k0 + cv * 8];
  }
  {
    const int row = tid >> 2, cv = tid & 3;
    *(v8u*)&Bs[row * BS + cv * 8] =
        *(const v8u*)&W[(nBase + row) * K + k0 + cv * 8];
    if (k0 + 32 < K)
      __builtin_prefetch(&W[(nBase + row) * K + k0 + 32 + cv * 8], 0, 1);
  }
#endif
}

// one BK=32 step: 2 A fragments, 4 B fragments, 8 WMMAs
__device__ __forceinline__ void compute_step(const unsigned short* As,
                                             const unsigned short* Bs, int wave,
                                             int h, int ln, v8f (&acc)[2][4]) {
  v16bf afrag[2];
#pragma unroll
  for (int a = 0; a < 2; ++a) {
    // A fragment (16x32): lane (ln, half h) holds K=[8h,8h+8) U [16+8h,+8)
    const int row = wave * 32 + a * 16 + ln;
    v8u lo = *(const v8u*)&As[row * AS + 8 * h];
    v8u hi = *(const v8u*)&As[row * AS + 16 + 8 * h];
    v16u au = __builtin_shufflevector(lo, hi, 0, 1, 2, 3, 4, 5, 6, 7, 8, 9, 10,
                                      11, 12, 13, 14, 15);
    afrag[a] = __builtin_bit_cast(v16bf, au);
  }
#pragma unroll
  for (int s = 0; s < 4; ++s) {
    // B fragment (32x16): lane (ln, half h) holds K=[8h,8h+8) U [16+8h,+8)
    const int row = s * 16 + ln;
    v8u blo = *(const v8u*)&Bs[row * BS + 8 * h];
    v8u bhi = *(const v8u*)&Bs[row * BS + 16 + 8 * h];
    v16u bu = __builtin_shufflevector(blo, bhi, 0, 1, 2, 3, 4, 5, 6, 7, 8, 9,
                                      10, 11, 12, 13, 14, 15);
    v16bf b = __builtin_bit_cast(v16bf, bu);
#pragma unroll
    for (int a = 0; a < 2; ++a)
      acc[a][s] = __builtin_amdgcn_wmma_f32_16x16x32_bf16(
          false, afrag[a], false, b, (short)0, acc[a][s], false, false);
  }
}

__global__ __launch_bounds__(256)
void gemm_wmma_bf16(const unsigned short* __restrict__ A,   // [M][K] bf16
                    const unsigned short* __restrict__ W,   // [N][K] bf16 ternary
                    const float* __restrict__ bias,         // [N]
                    float* __restrict__ outF,               // [M][N] f32 or null
                    unsigned short* __restrict__ outB,      // [M][N] bf16 or null
                    int M, int N, int K, int relu) {
  __shared__ unsigned short Abuf[2][256 * AS];  // 2 x 20 KB
  __shared__ unsigned short Bbuf[2][64 * BS];   // 2 x  5 KB

  const int tid  = threadIdx.x;
  const int wave = tid >> 5;
  const int lane = tid & 31;
  const int h    = lane >> 4;
  const int ln   = lane & 15;

  const size_t mBase = (size_t)blockIdx.y * 256;
  const size_t nBase = (size_t)blockIdx.x * 64;

  v8f acc[2][4] = {};

#if HAVE_ASYNC_LDS
  // ---- double-buffered async pipeline, last step peeled ----
  issue_tile(A, W, Abuf[0], Bbuf[0], mBase, nBase, K, 0, tid);
  int p = 0;
  for (int k0 = 0; k0 < K - 32; k0 += 32) {
    issue_tile(A, W, Abuf[1 - p], Bbuf[1 - p], mBase, nBase, K, k0 + 32, tid);
    ASYNC_WAIT(5);    // this wave's tile-k0 loads retired (in-order)
    __syncthreads();  // tile k0 visible to all waves
    compute_step(Abuf[p], Bbuf[p], wave, h, ln, acc);
    __syncthreads();  // all waves done reading buf p before it is rewritten
    p ^= 1;
  }
  ASYNC_WAIT(0);      // final tile complete
  __syncthreads();
  compute_step(Abuf[p], Bbuf[p], wave, h, ln, acc);
#else
  // ---- fallback: synchronous staging, single buffer ----
  for (int k0 = 0; k0 < K; k0 += 32) {
    issue_tile(A, W, Abuf[0], Bbuf[0], mBase, nBase, K, k0, tid);
    __syncthreads();
    compute_step(Abuf[0], Bbuf[0], wave, h, ln, acc);
    __syncthreads();
  }
#endif

  // ---- epilogue: C VGPR r -> row (r + 8h), col = ln ----
#pragma unroll
  for (int a = 0; a < 2; ++a) {
#pragma unroll
    for (int s = 0; s < 4; ++s) {
      const size_t col = nBase + s * 16 + ln;
      const float bv = bias[col];
#pragma unroll
      for (int r = 0; r < 8; ++r) {
        const size_t row = mBase + (size_t)wave * 32 + a * 16 + r + 8 * h;
        float c = acc[a][s][r] + bv;
        if (relu) c = fmaxf(c, 0.f);
        if (outB) outB[row * (size_t)N + col] = f32_to_bf16(c);
        else      outF[row * (size_t)N + col] = c;
      }
    }
  }
}

// ---------- launcher ----------
extern "C" void kernel_launch(void* const* d_in, const int* in_sizes, int n_in,
                              void* d_out, int out_size, void* d_ws,
                              size_t ws_size, hipStream_t stream) {
  (void)in_sizes; (void)n_in; (void)out_size; (void)ws_size;
  const float* X   = (const float*)d_in[0];
  const float* W1  = (const float*)d_in[1];
  const float* b1  = (const float*)d_in[2];
  const float* W2  = (const float*)d_in[3];
  const float* b2  = (const float*)d_in[4];
  const float* Wt1 = (const float*)d_in[5];
  const float* bt1 = (const float*)d_in[6];
  const float* Wt2 = (const float*)d_in[7];
  const float* bt2 = (const float*)d_in[8];
  float* out = (float*)d_out;

  const int ROWS = 8192, D_IN = 1024, D_H = 2048, D_V1 = 4096, D_V2 = 32000;

  unsigned char* ws = (unsigned char*)d_ws;
  size_t off = 0;
  auto carve = [&](size_t elems) {
    void* p = ws + off;
    off = (off + elems * sizeof(unsigned short) + 255) & ~(size_t)255;
    return (unsigned short*)p;
  };
  unsigned short* Xb   = carve((size_t)ROWS * D_IN);
  unsigned short* W1q  = carve((size_t)D_H  * D_IN);
  unsigned short* W2q  = carve((size_t)D_H  * D_H);
  unsigned short* Wt1q = carve((size_t)D_V1 * D_H);
  unsigned short* Wt2q = carve((size_t)D_V2 * D_H);  // 131 MB, L2-resident
  unsigned short* H1   = carve((size_t)ROWS * D_H);
  unsigned short* H2   = carve((size_t)ROWS * D_H);

  cvt_f32_bf16<<<1024, 256, 0, stream>>>(X, Xb, (size_t)ROWS * D_IN);

  ternary_quant_rows<<<D_H,  256, 0, stream>>>(W1,  W1q,  D_IN);
  ternary_quant_rows<<<D_H,  256, 0, stream>>>(W2,  W2q,  D_H);
  ternary_quant_rows<<<D_V1, 256, 0, stream>>>(Wt1, Wt1q, D_H);
  ternary_quant_rows<<<D_V2, 256, 0, stream>>>(Wt2, Wt2q, D_H);

  dim3 blk(256);
  gemm_wmma_bf16<<<dim3(D_H / 64,  ROWS / 256), blk, 0, stream>>>(
      Xb, W1q, b1, nullptr, H1, ROWS, D_H, D_IN, 1);
  gemm_wmma_bf16<<<dim3(D_H / 64,  ROWS / 256), blk, 0, stream>>>(
      H1, W2q, b2, nullptr, H2, ROWS, D_H, D_H, 1);
  gemm_wmma_bf16<<<dim3(D_V1 / 64, ROWS / 256), blk, 0, stream>>>(
      H2, Wt1q, bt1, out, nullptr, ROWS, D_V1, D_H, 0);
  gemm_wmma_bf16<<<dim3(D_V2 / 64, ROWS / 256), blk, 0, stream>>>(
      H2, Wt2q, bt2, out + (size_t)ROWS * D_V1, nullptr, ROWS, D_V2, D_H, 0);
}
